// QuadraticModelMultivariate_79370995630652
// MI455X (gfx1250) — matrix operand order repaired and checked by reading, not compiled, not run on hardware
//
#include <hip/hip_runtime.h>
#include <cstdint>

#define DEV __device__ __forceinline__

typedef __attribute__((ext_vector_type(16))) __bf16 v16bf;
typedef __attribute__((ext_vector_type(2)))  __bf16 v2bf;
typedef __attribute__((ext_vector_type(8)))  float  v8f;

constexpr int D      = 256;     // hidden width
constexpr int Q      = 33152;   // 256 + 32896 reference features
constexpr int NPAIR  = 32896;   // d*(d+1)/2
constexpr int NCHUNK = 4224;    // 8-aligned 8-wide triu chunks: sum(32 - i/8)
constexpr int KP     = 34048;   // 256 + 8*NCHUNK = 133*256, guard-free K
constexpr int BATCH  = 4096;

struct BF16Frag { uint4 lo, hi; };

DEV v16bf make_frag(uint4 lo, uint4 hi) {
  BF16Frag f{lo, hi};
  return __builtin_bit_cast(v16bf, f);
}

DEV uint32_t f2bf_bits(float a) {  // RNE f32 -> bf16 bits
  uint32_t u = __builtin_bit_cast(uint32_t, a);
  return (u + 0x7fffu + ((u >> 16) & 1u)) >> 16;
}

DEV uint32_t pack_bf16(float a, float b) {
  return f2bf_bits(a) | (f2bf_bits(b) << 16);
}

DEV uint32_t pkmul(uint32_t a, uint32_t b) {  // v_pk_mul_bf16
  v2bf x = __builtin_bit_cast(v2bf, a);
  v2bf y = __builtin_bit_cast(v2bf, b);
  v2bf r = x * y;
  return __builtin_bit_cast(uint32_t, r);
}

DEV v8f zero8() {
  v8f z;
#pragma unroll
  for (int i = 0; i < 8; ++i) z[i] = 0.f;
  return z;
}

// ---------------------------------------------------------------------------
// Chunk table: chunk c -> (i, j0), covering row i with 8-wide, 8-aligned j
// blocks from 8*(i/8) to 255.  chunks[c] = i | (j0 << 16).
// ---------------------------------------------------------------------------
__global__ void prep_chunks_kernel(uint32_t* __restrict__ chunks) {
  int i = threadIdx.x;  // 256 threads, one per row
  int G = i >> 3;
  int base = 32 * i - 4 * G * (G - 1) - (i & 7) * G;  // prefix sum of (32 - r/8)
  int n = 32 - G;
  for (int m = 0; m < n; ++m)
    chunks[base + m] = (uint32_t)i | ((uint32_t)(8 * (G + m)) << 16);
}

// ---------------------------------------------------------------------------
// Reference-order triu pair table (for the small output layer only).
// ---------------------------------------------------------------------------
__global__ void prep_pairs_kernel(uint16_t* __restrict__ pairs) {
  int i = threadIdx.x;
  int off = i * D - (i * (i - 1)) / 2;
  for (int j = i; j < D; ++j) pairs[off + (j - i)] = (uint16_t)(i | (j << 8));
}

// ---------------------------------------------------------------------------
// W (O x Q, f32) -> Wb (O x KP, bf16) permuted into chunk order; pad entries
// (j < i) and tail get zero weight.
// ---------------------------------------------------------------------------
__global__ void prep_w_kernel(const float* __restrict__ W,
                              uint16_t* __restrict__ Wb,
                              const uint32_t* __restrict__ chunks, int O) {
  long idx = (long)blockIdx.x * blockDim.x + threadIdx.x;
  long total = (long)O * KP;
  if (idx >= total) return;
  int o = (int)(idx / KP);
  int k = (int)(idx % KP);
  float v = 0.f;
  if (k < D) {
    v = W[(long)o * Q + k];
  } else {
    int c = (k - D) >> 3;
    uint32_t cd = chunks[c];
    int i = (int)(cd & 0xffffu);
    int j = (int)(cd >> 16) + ((k - D) & 7);
    if (j >= i) {
      int t = i * D - (i * (i - 1)) / 2 + (j - i);
      v = W[(long)o * Q + D + t];
    }
  }
  Wb[idx] = (uint16_t)f2bf_bits(v);
}

// ---------------------------------------------------------------------------
// Hidden layer: h_out(B x 256) = quadfeat(h_in) @ Wb^T + bias via bf16 WMMA.
// Grid: B/32 WGs of 256 threads (8 wave32s); wave w owns cols [32w, 32w+32).
// ---------------------------------------------------------------------------
__global__ __launch_bounds__(256) void layer_wmma_kernel(
    const float*    __restrict__ h_in,
    const uint16_t* __restrict__ Wb,
    const float*    __restrict__ bias,
    const uint32_t* __restrict__ chunks,
    float*          __restrict__ h_out) {
  constexpr int HBS  = 132;  // u32 stride: 128 packed-bf16 words + pad (528B)
  constexpr int ZSTR = 132;
  __shared__ uint32_t sh_hb[32 * HBS];   // 32 x 256 h in packed bf16, 16.9 KB
  __shared__ uint32_t zbuf[32 * ZSTR];   // 32 x 256 Z tile, packed bf16

  const int tid  = threadIdx.x;
  const int lane = tid & 31;
  const int wave = tid >> 5;
  const int row0 = blockIdx.x * 32;

  // Stage h tile as packed bf16 (coalesced b64 loads, RNE convert).
  for (int e = tid; e < 32 * 128; e += 256) {
    int r = e >> 7, c = e & 127;
    const float2 v = *(const float2*)&h_in[(row0 + r) * D + c * 2];
    sh_hb[r * HBS + c] = pack_bf16(v.x, v.y);
  }
  __syncthreads();

  // Z-generation assignment: thread -> (row, 32-wide k segment) = 4 chunks.
  const int zrow = tid >> 3;
  const int zseg = tid & 7;
  const uint32_t* hbr = &sh_hb[zrow * HBS];
  uint32_t*       zp  = &zbuf[zrow * ZSTR + zseg * 16];

  const int obase = wave * 32;
  const int laneN = lane & 15;
  const int hi    = lane >> 4;

  v8f acc[2][2];
#pragma unroll
  for (int mt = 0; mt < 2; ++mt)
#pragma unroll
    for (int nt = 0; nt < 2; ++nt) acc[mt][nt] = zero8();

  for (int kb = 0; kb < KP; kb += 256) {
    // ---- cooperative Z tile generation (packed bf16 in LDS) ----
    if (kb == 0) {              // linear features: straight copy of bf16 h
#pragma unroll
      for (int u = 0; u < 4; ++u)
        *(uint4*)(zp + u * 4) = *(const uint4*)&hbr[zseg * 16 + u * 4];
    } else {                    // quad: z[8] = h[i] * h[j0..j0+7], j0 % 8 == 0
      int cbase = ((kb - 256) >> 3) + zseg * 4;
#pragma unroll
      for (int cc = 0; cc < 4; ++cc) {
        uint32_t cd  = chunks[cbase + cc];
        uint32_t ii  = cd & 0xffffu;
        uint32_t j0  = cd >> 16;
        uint32_t hw  = hbr[ii >> 1];
        uint32_t h16 = (hw >> ((ii & 1u) << 4)) & 0xffffu;
        uint32_t hs  = h16 | (h16 << 16);             // splat bf16 h[i]
        const uint4 hj = *(const uint4*)&hbr[j0 >> 1]; // ds_load_b128, aligned
        uint4 r;
        r.x = pkmul(hs, hj.x);
        r.y = pkmul(hs, hj.y);
        r.z = pkmul(hs, hj.z);
        r.w = pkmul(hs, hj.w);
        *(uint4*)(zp + cc * 4) = r;                    // ds_store_b128
      }
    }
    __syncthreads();

    // ---- GEMM over this 256-wide K block, K=32 per WMMA ----
#pragma unroll 2
    for (int kk = 0; kk < 256; kk += 32) {
      v16bf afrag[2];
#pragma unroll
      for (int mt = 0; mt < 2; ++mt) {
        const uint32_t* ap =
            &zbuf[(mt * 16 + laneN) * ZSTR + ((kk + hi * 8) >> 1)];
        uint4 alo = *(const uint4*)ap;
        uint4 ahi = *(const uint4*)(ap + 8);
        afrag[mt] = make_frag(alo, ahi);
      }
      v16bf bfrag[2];
#pragma unroll
      for (int nt = 0; nt < 2; ++nt) {
        int o = obase + nt * 16 + laneN;
        const uint4* bp =
            (const uint4*)(Wb + (size_t)o * KP + kb + kk + hi * 8);
        uint4 blo = bp[0];
        uint4 bhi = bp[2];
        __builtin_prefetch(bp + 4, 0, 1);
        bfrag[nt] = make_frag(blo, bhi);
      }
#pragma unroll
      for (int mt = 0; mt < 2; ++mt)
#pragma unroll
        for (int nt = 0; nt < 2; ++nt)
          acc[mt][nt] = __builtin_amdgcn_wmma_f32_16x16x32_bf16(
              false, afrag[mt], false, bfrag[nt], (short)0, acc[mt][nt],
              false, false);
    }
    __syncthreads();
  }

  // Epilogue: lane n = N, VGPR r: M = 16*mt + 8*hi + r.
#pragma unroll
  for (int nt = 0; nt < 2; ++nt) {
    int o = obase + nt * 16 + laneN;
    float bv = bias[o];
#pragma unroll
    for (int mt = 0; mt < 2; ++mt) {
      int rbase = row0 + mt * 16 + hi * 8;
#pragma unroll
      for (int r = 0; r < 8; ++r)
        h_out[(rbase + r) * D + o] = acc[mt][nt][r] + bv;
    }
  }
}

// ---------------------------------------------------------------------------
// Output layer (O=10, ~2% of FLOPs): plain VALU with h tile in LDS.
// ---------------------------------------------------------------------------
__global__ __launch_bounds__(256) void layer_out_kernel(
    const float*    __restrict__ h_in,
    const float*    __restrict__ W2,   // 10 x Q, f32 (reference order)
    const float*    __restrict__ b2,
    const uint16_t* __restrict__ pairs,
    float*          __restrict__ out) {
  constexpr int HSTR = D + 4;
  __shared__ float sh[16 * HSTR];
  const int tid  = threadIdx.x;
  const int row0 = blockIdx.x * 16;
  for (int e = tid; e < 16 * D; e += 256) {
    int r = e >> 8, c = e & 255;
    sh[r * HSTR + c] = h_in[(row0 + r) * D + c];
  }
  __syncthreads();

  const int o = tid & 15;
  const int r = tid >> 4;
  if (o >= 10) return;
  const float* hr = &sh[r * HSTR];
  const float* wr = W2 + (size_t)o * Q;
  float acc = b2[o];
#pragma unroll 4
  for (int k = 0; k < D; ++k) acc += hr[k] * wr[k];
  const float* wq = wr + D;
#pragma unroll 4
  for (int t = 0; t < NPAIR; ++t) {
    uint32_t p = pairs[t];
    acc += hr[p & 255] * hr[p >> 8] * wq[t];
  }
  out[(row0 + r) * 10 + o] = acc;
}

// ---------------------------------------------------------------------------
extern "C" void kernel_launch(void* const* d_in, const int* in_sizes, int n_in,
                              void* d_out, int out_size, void* d_ws,
                              size_t ws_size, hipStream_t stream) {
  (void)in_sizes; (void)n_in; (void)out_size; (void)ws_size;
  const float* x  = (const float*)d_in[0];
  const float* W0 = (const float*)d_in[1];
  const float* b0 = (const float*)d_in[2];
  const float* W1 = (const float*)d_in[3];
  const float* b1 = (const float*)d_in[4];
  const float* W2 = (const float*)d_in[5];
  const float* b2 = (const float*)d_in[6];
  float* out = (float*)d_out;

  // Workspace layout (~41.4 MB):
  //   chunks: 4224 * 4B        @ 0
  //   pairs:  32896 * 2B       @ 16896
  //   Wb0:    256*KP*2B        @ 82944
  //   Wb1:    256*KP*2B
  //   h1, h2: 4096*256 f32 each
  char* ws = (char*)d_ws;
  uint32_t* chunks = (uint32_t*)ws;
  uint16_t* pairs  = (uint16_t*)(ws + 16896);
  size_t    wbytes = (size_t)256 * KP * 2;
  uint16_t* Wb0 = (uint16_t*)(ws + 82944);
  uint16_t* Wb1 = (uint16_t*)(ws + 82944 + wbytes);
  float*    h1  = (float*)(ws + 82944 + 2 * wbytes);
  float*    h2  = h1 + (size_t)BATCH * D;

  hipLaunchKernelGGL(prep_chunks_kernel, dim3(1), dim3(256), 0, stream, chunks);
  hipLaunchKernelGGL(prep_pairs_kernel, dim3(1), dim3(256), 0, stream, pairs);
  const int wblocks = (256 * KP) / 256;  // one thread per Wb element
  hipLaunchKernelGGL(prep_w_kernel, dim3(wblocks), dim3(256), 0, stream,
                     W0, Wb0, chunks, 256);
  hipLaunchKernelGGL(prep_w_kernel, dim3(wblocks), dim3(256), 0, stream,
                     W1, Wb1, chunks, 256);

  hipLaunchKernelGGL(layer_wmma_kernel, dim3(BATCH / 32), dim3(256), 0, stream,
                     x, Wb0, b0, chunks, h1);
  hipLaunchKernelGGL(layer_wmma_kernel, dim3(BATCH / 32), dim3(256), 0, stream,
                     h1, Wb1, b1, chunks, h2);
  hipLaunchKernelGGL(layer_out_kernel, dim3(BATCH / 16), dim3(256), 0, stream,
                     h2, W2, b2, pairs, out);
}
